// AstraloraLayer_58205396796140
// MI455X (gfx1250) — 8-bit matrix operand order repaired and checked by reading, compiled, not run on hardware
//
#include <hip/hip_runtime.h>

typedef int v8i __attribute__((ext_vector_type(8)));

#define D_INP 1024
#define D_OUT 1024
#define N_ROWS (16 * 2048)

#define BM 128
#define BN 128
#define BK 64
#define KTILES (D_INP / BK)

__constant__ const float kXMIN = -3.0f;
__constant__ const float kWMIN = -0.2f;
#define X_STEP (6.0f / 255.0f)
#define W_STEP (0.4f / 255.0f)

__device__ __forceinline__ void async_load_b128(unsigned lds_addr,
                                                const void* gaddr) {
  asm volatile("global_load_async_to_lds_b128 %0, %1, off"
               :
               : "v"(lds_addr), "v"(gaddr)
               : "memory");
}

__device__ __forceinline__ void wait_asynccnt0() {
#if __has_builtin(__builtin_amdgcn_s_wait_asynccnt)
  __builtin_amdgcn_s_wait_asynccnt(0);
#else
  asm volatile("s_wait_asynccnt 0" ::: "memory");
#endif
}

__global__ __launch_bounds__(256) void quant_rows_kernel(
    const float* __restrict__ src, unsigned char* __restrict__ codes,
    int* __restrict__ rsum, float vmin, float vmax, float inv_step) {
  __shared__ int red[256];
  const int row = blockIdx.x;
  const int t = threadIdx.x;

  const float4 v = ((const float4*)(src + (size_t)row * D_INP))[t];

  int k0 = (int)rintf((fminf(fmaxf(v.x, vmin), vmax) - vmin) * inv_step);
  int k1 = (int)rintf((fminf(fmaxf(v.y, vmin), vmax) - vmin) * inv_step);
  int k2 = (int)rintf((fminf(fmaxf(v.z, vmin), vmax) - vmin) * inv_step);
  int k3 = (int)rintf((fminf(fmaxf(v.w, vmin), vmax) - vmin) * inv_step);
  k0 = min(max(k0, 0), 255);
  k1 = min(max(k1, 0), 255);
  k2 = min(max(k2, 0), 255);
  k3 = min(max(k3, 0), 255);

  unsigned int packed = (unsigned int)k0 | ((unsigned int)k1 << 8) |
                        ((unsigned int)k2 << 16) | ((unsigned int)k3 << 24);
  ((unsigned int*)(codes + (size_t)row * D_INP))[t] = packed;

  red[t] = k0 + k1 + k2 + k3;
  __syncthreads();
#pragma unroll
  for (int off = 128; off > 0; off >>= 1) {
    if (t < off) red[t] += red[t + off];
    __syncthreads();
  }
  if (t == 0) rsum[row] = red[0];
}

__device__ __forceinline__ void compute_tile(
    const unsigned char (&A)[BM][BK], const unsigned char (&B)[BN][BK],
    int wave_m, int wave_n, int half, int l16, v8i (&acc)[4][2]) {
  v8i a[4], b[2];
#pragma unroll
  for (int mi = 0; mi < 4; ++mi) {
    const unsigned char* p = &A[wave_m * 64 + mi * 16 + l16][half * 8];
    union { uint2 d[4]; v8i v; } fa;
    fa.d[0] = *(const uint2*)(p + 0);
    fa.d[1] = *(const uint2*)(p + 16);
    fa.d[2] = *(const uint2*)(p + 32);
    fa.d[3] = *(const uint2*)(p + 48);
    a[mi] = fa.v;
  }
#pragma unroll
  for (int ni = 0; ni < 2; ++ni) {
    const unsigned char* p = &B[wave_n * 32 + ni * 16 + l16][half * 8];
    union { uint2 q[4]; v8i v; } fb;
    fb.q[0] = *(const uint2*)(p + 0);
    fb.q[1] = *(const uint2*)(p + 16);
    fb.q[2] = *(const uint2*)(p + 32);
    fb.q[3] = *(const uint2*)(p + 48);
    b[ni] = fb.v;
  }
#pragma unroll
  for (int mi = 0; mi < 4; ++mi)
#pragma unroll
    for (int ni = 0; ni < 2; ++ni)
      acc[mi][ni] = __builtin_amdgcn_wmma_i32_16x16x64_iu8(
           false, a[mi],  false, b[ni], acc[mi][ni],
           false,  false);
}

__global__ __launch_bounds__(256) void gemm_iu8_kernel(
    const unsigned char* __restrict__ xq, const unsigned char* __restrict__ wq,
    const int* __restrict__ xsum, const int* __restrict__ wsum,
    const float* __restrict__ scale, float* __restrict__ out) {
  __shared__ __align__(16) unsigned char As[2][BM][BK];
  __shared__ __align__(16) unsigned char Bs[2][BN][BK];

  const int tid = threadIdx.x;
  const int lane = tid & 31;
  const int wid = tid >> 5;
  const int wave_m = wid >> 2;
  const int wave_n = wid & 3;
  const int half = lane >> 4;
  const int l16 = lane & 15;

  const int o0 = blockIdx.x * BN;
  const int n0 = blockIdx.y * BM;

  const int row0 = tid >> 2;
  const int row1 = row0 + 64;
  const int c0 = (tid & 3) * 16;

  const unsigned char* gA0 = xq + (size_t)(n0 + row0) * D_INP + c0;
  const unsigned char* gA1 = xq + (size_t)(n0 + row1) * D_INP + c0;
  const unsigned char* gB0 = wq + (size_t)(o0 + row0) * D_INP + c0;
  const unsigned char* gB1 = wq + (size_t)(o0 + row1) * D_INP + c0;

  unsigned lA0[2], lA1[2], lB0[2], lB1[2];
#pragma unroll
  for (int b = 0; b < 2; ++b) {
    lA0[b] = (unsigned)(uintptr_t)&As[b][row0][c0];
    lA1[b] = (unsigned)(uintptr_t)&As[b][row1][c0];
    lB0[b] = (unsigned)(uintptr_t)&Bs[b][row0][c0];
    lB1[b] = (unsigned)(uintptr_t)&Bs[b][row1][c0];
  }

  v8i acc[4][2];
#pragma unroll
  for (int mi = 0; mi < 4; ++mi)
#pragma unroll
    for (int ni = 0; ni < 2; ++ni) acc[mi][ni] = (v8i){0, 0, 0, 0, 0, 0, 0, 0};

  async_load_b128(lA0[0], gA0);
  async_load_b128(lA1[0], gA1);
  async_load_b128(lB0[0], gB0);
  async_load_b128(lB1[0], gB1);
  gA0 += BK; gA1 += BK; gB0 += BK; gB1 += BK;
  wait_asynccnt0();
  __syncthreads();

#pragma unroll 1
  for (int kt = 0; kt < KTILES; kt += 2) {
    {
      async_load_b128(lA0[1], gA0);
      async_load_b128(lA1[1], gA1);
      async_load_b128(lB0[1], gB0);
      async_load_b128(lB1[1], gB1);
      gA0 += BK; gA1 += BK; gB0 += BK; gB1 += BK;
    }
    compute_tile(As[0], Bs[0], wave_m, wave_n, half, l16, acc);
    wait_asynccnt0();
    __syncthreads();

    if (kt + 2 < KTILES) {
      async_load_b128(lA0[0], gA0);
      async_load_b128(lA1[0], gA1);
      async_load_b128(lB0[0], gB0);
      async_load_b128(lB1[0], gB1);
      gA0 += BK; gA1 += BK; gB0 += BK; gB1 += BK;
    }
    compute_tile(As[1], Bs[1], wave_m, wave_n, half, l16, acc);
    wait_asynccnt0();
    __syncthreads();
  }

  const float scl = scale[0];
  const float sxw = X_STEP * W_STEP;
  const float c1 = kXMIN * W_STEP;
  const float c2 = kWMIN * X_STEP;
  const float c3 = (float)D_INP * kXMIN * kWMIN;

#pragma unroll
  for (int ni = 0; ni < 2; ++ni) {
    const int o = o0 + wave_n * 32 + ni * 16 + l16;
    const float t_o = c1 * (float)wsum[o] + c3;
#pragma unroll
    for (int mi = 0; mi < 4; ++mi) {
      const int nbase = n0 + wave_m * 64 + mi * 16 + half * 8;
#pragma unroll
      for (int v = 0; v < 8; ++v) {
        const int n = nbase + v;
        const float y =
            sxw * (float)acc[mi][ni][v] + t_o + c2 * (float)xsum[n];
        out[(size_t)n * D_OUT + o] = y * scl;
      }
    }
  }
}

extern "C" void kernel_launch(void* const* d_in, const int* in_sizes, int n_in,
                              void* d_out, int out_size, void* d_ws,
                              size_t ws_size, hipStream_t stream) {
  const float* x = (const float*)d_in[0];
  const float* w = (const float*)d_in[1];
  const float* scale = (const float*)d_in[2];
  float* out = (float*)d_out;

  unsigned char* ws = (unsigned char*)d_ws;
  unsigned char* xq = ws;
  unsigned char* wq = ws + (size_t)N_ROWS * D_INP;
  int* xsum = (int*)(wq + (size_t)D_OUT * D_INP);
  int* wsum = xsum + N_ROWS;

  quant_rows_kernel<<<N_ROWS, 256, 0, stream>>>(x, xq, xsum, -3.0f, 3.0f,
                                                255.0f / 6.0f);
  quant_rows_kernel<<<D_OUT, 256, 0, stream>>>(w, wq, wsum, -0.2f, 0.2f,
                                               255.0f / 0.4f);

  dim3 grid(D_OUT / BN, N_ROWS / BM);
  gemm_iu8_kernel<<<grid, 256, 0, stream>>>(xq, wq, xsum, wsum, scale, out);
}
